// SparseTransformer_16956530885277
// MI455X (gfx1250) — matrix operand order repaired and checked by reading, not compiled, not run on hardware
//
#include <hip/hip_runtime.h>
#include <hip/hip_bf16.h>

// ---- model dims ----
#define NN 2048
#define DD 1024
#define HH 16
#define DHD 64
#define FF 4096
#define VV 32000
#define LL 4
#define BSB 128
#define WINW 4

typedef __attribute__((ext_vector_type(16))) __bf16 v16bf;
typedef __attribute__((ext_vector_type(8)))  float  v8f;

union Frag16 { v16bf v; uint4 q[2]; };

static __device__ __forceinline__ unsigned short f2bf(float f) {
  union { float f; unsigned u; } x; x.f = f;
  unsigned r = x.u + 0x7FFFu + ((x.u >> 16) & 1u);
  return (unsigned short)(r >> 16);
}
static __device__ __forceinline__ v8f vzero() {
  v8f z; for (int i = 0; i < 8; ++i) z[i] = 0.f; return z;
}

// ------------------------------------------------------------------
// CDNA5 async global->LDS copy (ASYNCcnt-tracked), 16 bytes per lane.
// Probe round 2 showed the builtin exists with signature
//   (v4i __device__*, v4i __shared__*, imm offset, imm cpol)
// where v4i = int __attribute__((vector_size(16))).
// ------------------------------------------------------------------
#if defined(__gfx1250__)
  #if __has_builtin(__builtin_amdgcn_global_load_async_to_lds_b128)
    #define ASYNC_MODE 2
  #else
    #define ASYNC_MODE 1
  #endif
#else
  #define ASYNC_MODE 0
#endif

typedef int v4i __attribute__((vector_size(16)));
typedef __attribute__((address_space(1))) v4i as1_v4i;
typedef __attribute__((address_space(3))) v4i as3_v4i;
typedef __attribute__((address_space(3))) unsigned short as3_ushort;

static __device__ __forceinline__ void async_copy16(const unsigned short* g,
                                                    unsigned short* l) {
#if ASYNC_MODE == 2
  __builtin_amdgcn_global_load_async_to_lds_b128((as1_v4i*)g, (as3_v4i*)l, 0, 0);
#elif ASYNC_MODE == 1
  unsigned lds = (unsigned)(size_t)(as3_ushort*)l;
  asm volatile("global_load_async_to_lds_b128 %0, %1, off"
               :: "v"(lds), "v"((unsigned long long)(size_t)g)
               : "memory");
#else
  *(uint4*)l = *(const uint4*)g;
#endif
}

static __device__ __forceinline__ void async_wait0() {
#if ASYNC_MODE == 2
  #if __has_builtin(__builtin_amdgcn_s_wait_asynccnt)
    __builtin_amdgcn_s_wait_asynccnt(0);
  #else
    asm volatile("s_wait_asynccnt 0x0" ::: "memory");
  #endif
#elif ASYNC_MODE == 1
  asm volatile("s_wait_asynccnt 0x0" ::: "memory");
#endif
}

// ------------------------------------------------------------------
// Weight convert + transpose: W[K][Nc] f32  ->  Wt[Nc][K] bf16
// ------------------------------------------------------------------
__global__ __launch_bounds__(256) void transpose_f32_to_bf16(
    const float* __restrict__ W, unsigned short* __restrict__ Wt, int K, int Nc) {
  __shared__ float tile[32][33];
  int n0 = blockIdx.x * 32, k0 = blockIdx.y * 32;
  int tx = threadIdx.x & 31, ty = threadIdx.x >> 5;   // ty: 0..7
  for (int i = 0; i < 32; i += 8)
    tile[ty + i][tx] = W[(size_t)(k0 + ty + i) * Nc + (n0 + tx)];
  __syncthreads();
  for (int i = 0; i < 32; i += 8)
    Wt[(size_t)(n0 + ty + i) * K + (k0 + tx)] = f2bf(tile[tx][ty + i]);
}

// ------------------------------------------------------------------
// Embedding: x[n][d] = tok[ids[n]][d] + pos[n][d]
// ------------------------------------------------------------------
__global__ __launch_bounds__(256) void embed_kernel(
    const int* __restrict__ ids, const float* __restrict__ tok,
    const float* __restrict__ pos, float* __restrict__ x) {
  int n = blockIdx.x;
  int id = ids[n];
  for (int d = threadIdx.x; d < DD; d += 256)
    x[(size_t)n * DD + d] = tok[(size_t)id * DD + d] + pos[(size_t)n * DD + d];
}

// ------------------------------------------------------------------
// LayerNorm fp32 -> bf16, one row per block (256 threads, 4 elems each)
// ------------------------------------------------------------------
__global__ __launch_bounds__(256) void layernorm_bf16(
    const float* __restrict__ x, const float* __restrict__ s,
    const float* __restrict__ b, unsigned short* __restrict__ h) {
  __shared__ float rs[8], rq[8];
  int row = blockIdx.x;
  const float* xr = x + (size_t)row * DD;
  float vals[4], lsum = 0.f, lsq = 0.f;
  for (int i = 0; i < 4; ++i) {
    float v = xr[threadIdx.x * 4 + i];
    vals[i] = v; lsum += v; lsq += v * v;
  }
  for (int m = 16; m >= 1; m >>= 1) {
    lsum += __shfl_xor(lsum, m, 32);
    lsq  += __shfl_xor(lsq,  m, 32);
  }
  if ((threadIdx.x & 31) == 0) { rs[threadIdx.x >> 5] = lsum; rq[threadIdx.x >> 5] = lsq; }
  __syncthreads();
  if (threadIdx.x == 0) {
    float a = 0.f, c = 0.f;
    for (int i = 0; i < 8; ++i) { a += rs[i]; c += rq[i]; }
    rs[0] = a; rq[0] = c;
  }
  __syncthreads();
  float mean = rs[0] * (1.f / DD);
  float var  = rq[0] * (1.f / DD) - mean * mean;
  float inv  = rsqrtf(var + 1e-6f);
  for (int i = 0; i < 4; ++i) {
    int d = threadIdx.x * 4 + i;
    h[(size_t)row * DD + d] = f2bf((vals[i] - mean) * inv * s[d] + b[d]);
  }
}

// ------------------------------------------------------------------
// GEMM: C[M][Nc] = A[M][K](bf16) x Bt[Nc][K](bf16) + bias, via WMMA bf16.
// Double-buffered LDS tiles staged with async global->LDS copies.
// MODE 0: Cf = acc+bias (f32)        MODE 1: Cb = bf16(acc+bias)
// MODE 2: Cb = bf16(gelu(acc+bias))  MODE 3: Cf += acc+bias (residual)
// Block: 256 thr (8 waves); tile 128x128, BK=64; wave -> 16x128 strip.
// ------------------------------------------------------------------
#define ASTR 72   // padded LDS row stride (elems) -> bank-conflict-free frags
template <int MODE>
__global__ __launch_bounds__(256) void gemm_bf16(
    const unsigned short* __restrict__ A, const unsigned short* __restrict__ Bt,
    const float* __restrict__ bias, float* __restrict__ Cf,
    unsigned short* __restrict__ Cb, int M, int Nc, int K) {
  __shared__ unsigned short As[2][128 * ASTR];
  __shared__ unsigned short Bs[2][128 * ASTR];
  int bm = blockIdx.y * 128, bn = blockIdx.x * 128;
  int tid = threadIdx.x;
  int wave = tid >> 5, lane = tid & 31;
  int laneLow = lane & 15, laneHalf = lane >> 4;
  int mw = wave * 16;

  v8f acc[8];
  for (int s = 0; s < 8; ++s) acc[s] = vzero();

  int ldr = tid >> 1, ldc = (tid & 1) * 32;  // each thread: 32 elems (64B) per tile
  const unsigned short* Ag = A  + (size_t)(bm + ldr) * K + ldc;
  const unsigned short* Bg = Bt + (size_t)(bn + ldr) * K + ldc;

  auto stage = [&](int buf, int k0) {
    const unsigned short* ga = Ag + k0;
    const unsigned short* gb = Bg + k0;
    unsigned short* la = &As[buf][ldr * ASTR + ldc];
    unsigned short* lb = &Bs[buf][ldr * ASTR + ldc];
#pragma unroll
    for (int i = 0; i < 4; ++i) {
      async_copy16(ga + i * 8, la + i * 8);
      async_copy16(gb + i * 8, lb + i * 8);
    }
  };

  stage(0, 0);
  async_wait0();
  __syncthreads();

  int cur = 0;
  for (int k0 = 0; k0 < K; k0 += 64) {
    if (k0 + 64 < K) {          // stage next tile while computing current
      stage(cur ^ 1, k0 + 64);
      __builtin_prefetch(Ag + k0 + 128, 0, 3);
      __builtin_prefetch(Bg + k0 + 128, 0, 3);
    }
    for (int kc = 0; kc < 2; ++kc) {
      Frag16 af;
      const unsigned short* ab = &As[cur][(mw + laneLow) * ASTR + kc * 32 + laneHalf * 8];
      af.q[0] = *(const uint4*)(ab);
      af.q[1] = *(const uint4*)(ab + 16);
      for (int s = 0; s < 8; ++s) {
        Frag16 bf;
        const unsigned short* bb = &Bs[cur][(s * 16 + laneLow) * ASTR + kc * 32 + laneHalf * 8];
        bf.q[0] = *(const uint4*)(bb);
        bf.q[1] = *(const uint4*)(bb + 16);
        acc[s] = __builtin_amdgcn_wmma_f32_16x16x32_bf16(
            false, af.v, false, bf.v, (short)0, acc[s], false, false);
      }
    }
    async_wait0();
    __syncthreads();
    cur ^= 1;
  }

  for (int s = 0; s < 8; ++s) {
    int col = bn + s * 16 + laneLow;
    float bb = bias ? bias[col] : 0.f;
    for (int r = 0; r < 8; ++r) {
      int row = bm + mw + r + laneHalf * 8;
      float v = acc[s][r] + bb;
      size_t idx = (size_t)row * Nc + col;
      if (MODE == 0) {
        Cf[idx] = v;
      } else if (MODE == 1) {
        Cb[idx] = f2bf(v);
      } else if (MODE == 2) {
        float g = 0.5f * v * (1.f + tanhf(0.7978845608f * (v + 0.044715f * v * v * v)));
        Cb[idx] = f2bf(g);
      } else {
        Cf[idx] += v;
      }
    }
  }
}

// ------------------------------------------------------------------
// Block-sparse causal local-window attention (flash-style online softmax).
// Grid: (H, N/128). Block: 256 thr = 8 waves; wave owns 16 query rows.
// ------------------------------------------------------------------
#define KSTR 72
#define VSTR 136
#define PSTR 136
__global__ __launch_bounds__(256) void attention_kernel(
    const unsigned short* __restrict__ q, const unsigned short* __restrict__ k,
    const unsigned short* __restrict__ v, unsigned short* __restrict__ out) {
  __shared__ unsigned short Ks[128 * KSTR];   // [key][dim]
  __shared__ unsigned short Vt[64 * VSTR];    // [dim][key]
  __shared__ unsigned short Ps[8 * 16 * PSTR];// per-wave P strips
  int h = blockIdx.x, qb = blockIdx.y;
  int tid = threadIdx.x, wave = tid >> 5, lane = tid & 31;
  int laneLow = lane & 15, laneHalf = lane >> 4;

  // Q fragments (16x64 strip per wave, straight from global: A-layout chunks)
  Frag16 qf[2];
  {
    int qrow = qb * 128 + wave * 16 + laneLow;
    const unsigned short* qbase = q + (size_t)qrow * DD + h * DHD;
    for (int kc = 0; kc < 2; ++kc) {
      const unsigned short* p = qbase + kc * 32 + laneHalf * 8;
      qf[kc].q[0] = *(const uint4*)(p);
      qf[kc].q[1] = *(const uint4*)(p + 16);
    }
  }

  v8f o[4];
  for (int s = 0; s < 4; ++s) o[s] = vzero();
  float mrow[8], lrow[8];
  for (int r = 0; r < 8; ++r) { mrow[r] = -1e30f; lrow[r] = 0.f; }

  for (int j = 0; j < WINW; ++j) {
    int raw = qb - (WINW - 1) + j;
    if (raw < 0) continue;           // uniform across block
    __syncthreads();
    { // stage K (natural, async) and V (transposed, scatter) into LDS
      int r = tid >> 1, c = (tid & 1) * 32;
      const unsigned short* ksrc = k + (size_t)(raw * 128 + r) * DD + h * DHD + c;
      unsigned short* kdst = Ks + r * KSTR + c;
#pragma unroll
      for (int i = 0; i < 4; ++i) async_copy16(ksrc + i * 8, kdst + i * 8);
      const unsigned short* vsrc = v + (size_t)(raw * 128 + r) * DD + h * DHD + c;
      for (int d = 0; d < 32; ++d) Vt[(c + d) * VSTR + r] = vsrc[d];
    }
    async_wait0();
    __syncthreads();

    // S = Q K^T   (16 x 128 per wave)
    v8f sacc[8];
    for (int s = 0; s < 8; ++s) sacc[s] = vzero();
    for (int kc = 0; kc < 2; ++kc) {
      for (int s = 0; s < 8; ++s) {
        Frag16 bf;
        const unsigned short* bb = Ks + (s * 16 + laneLow) * KSTR + kc * 32 + laneHalf * 8;
        bf.q[0] = *(const uint4*)(bb);
        bf.q[1] = *(const uint4*)(bb + 16);
        sacc[s] = __builtin_amdgcn_wmma_f32_16x16x32_bf16(
            false, qf[kc].v, false, bf.v, (short)0, sacc[s], false, false);
      }
    }

    const bool diag = (raw == qb);
    // scale + mask + row max (cross-lane over 16-lane halves)
    float pm[8];
    for (int r = 0; r < 8; ++r) {
      int rowLocal = wave * 16 + r + laneHalf * 8;
      float mx = -1e30f;
      for (int s = 0; s < 8; ++s) {
        float val = sacc[s][r] * 0.125f;
        int colLocal = s * 16 + laneLow;
        if (diag && colLocal > rowLocal) val = -1e9f;
        sacc[s][r] = val;
        mx = fmaxf(mx, val);
      }
      for (int m = 8; m >= 1; m >>= 1) mx = fmaxf(mx, __shfl_xor(mx, m, 32));
      pm[r] = mx;
    }
    // online softmax update
    for (int r = 0; r < 8; ++r) {
      float mnew = fmaxf(mrow[r], pm[r]);
      float sc = __expf(mrow[r] - mnew);
      float rsum = 0.f;
      for (int s = 0; s < 8; ++s) {
        float p = __expf(sacc[s][r] - mnew);
        sacc[s][r] = p;
        rsum += p;
      }
      for (int m = 8; m >= 1; m >>= 1) rsum += __shfl_xor(rsum, m, 32);
      lrow[r] = lrow[r] * sc + rsum;
      mrow[r] = mnew;
      for (int s = 0; s < 4; ++s) o[s][r] *= sc;
    }
    // round-trip P through per-wave LDS region to re-fragment as A-layout
    unsigned short* pw = Ps + wave * 16 * PSTR;
    for (int s = 0; s < 8; ++s)
      for (int r = 0; r < 8; ++r)
        pw[(r + laneHalf * 8) * PSTR + s * 16 + laneLow] = f2bf(sacc[s][r]);
    // O += P V
    for (int kc = 0; kc < 4; ++kc) {
      Frag16 af;
      const unsigned short* ab = pw + laneLow * PSTR + kc * 32 + laneHalf * 8;
      af.q[0] = *(const uint4*)(ab);
      af.q[1] = *(const uint4*)(ab + 16);
      for (int s = 0; s < 4; ++s) {
        Frag16 bf;
        const unsigned short* bb = Vt + (s * 16 + laneLow) * VSTR + kc * 32 + laneHalf * 8;
        bf.q[0] = *(const uint4*)(bb);
        bf.q[1] = *(const uint4*)(bb + 16);
        o[s] = __builtin_amdgcn_wmma_f32_16x16x32_bf16(
            false, af.v, false, bf.v, (short)0, o[s], false, false);
      }
    }
  }

  for (int s = 0; s < 4; ++s)
    for (int r = 0; r < 8; ++r) {
      int token = qb * 128 + wave * 16 + r + laneHalf * 8;
      int dcol = h * DHD + s * 16 + laneLow;
      out[(size_t)token * DD + dcol] = f2bf(o[s][r] / lrow[r]);
    }
}

// ------------------------------------------------------------------
// Host-side orchestration
// ------------------------------------------------------------------
extern "C" void kernel_launch(void* const* d_in, const int* in_sizes, int n_in,
                              void* d_out, int out_size, void* d_ws, size_t ws_size,
                              hipStream_t stream) {
  (void)in_sizes; (void)n_in; (void)out_size; (void)ws_size;
  const int*   ids   = (const int*)  d_in[0];
  const float* tok   = (const float*)d_in[1];
  const float* pos   = (const float*)d_in[2];
  const float* ln1_s = (const float*)d_in[3];
  const float* ln1_b = (const float*)d_in[4];
  const float* Wq    = (const float*)d_in[5];
  const float* bq    = (const float*)d_in[6];
  const float* Wk    = (const float*)d_in[7];
  const float* bk    = (const float*)d_in[8];
  const float* Wv    = (const float*)d_in[9];
  const float* bv    = (const float*)d_in[10];
  const float* Wo    = (const float*)d_in[11];
  const float* bo    = (const float*)d_in[12];
  const float* ln2_s = (const float*)d_in[13];
  const float* ln2_b = (const float*)d_in[14];
  const float* Wup   = (const float*)d_in[15];
  const float* bup   = (const float*)d_in[16];
  const float* Wdown = (const float*)d_in[17];
  const float* bdown = (const float*)d_in[18];
  const float* fin_s = (const float*)d_in[19];
  const float* fin_b = (const float*)d_in[20];
  const float* Wlm   = (const float*)d_in[21];
  const float* blm   = (const float*)d_in[22];

  // workspace carve
  char* w = (char*)d_ws;
  auto carve = [&](size_t bytes) {
    void* p = (void*)w;
    w += (bytes + 255) & ~(size_t)255;
    return p;
  };
  unsigned short* WqT = (unsigned short*)carve((size_t)LL * DD * DD * 2);
  unsigned short* WkT = (unsigned short*)carve((size_t)LL * DD * DD * 2);
  unsigned short* WvT = (unsigned short*)carve((size_t)LL * DD * DD * 2);
  unsigned short* WoT = (unsigned short*)carve((size_t)LL * DD * DD * 2);
  unsigned short* WupT   = (unsigned short*)carve((size_t)LL * DD * FF * 2);
  unsigned short* WdownT = (unsigned short*)carve((size_t)LL * DD * FF * 2);
  unsigned short* WlmT   = (unsigned short*)carve((size_t)DD * VV * 2);
  float*          xbuf   = (float*)         carve((size_t)NN * DD * 4);
  unsigned short* hbuf   = (unsigned short*)carve((size_t)NN * DD * 2);
  unsigned short* qbuf   = (unsigned short*)carve((size_t)NN * DD * 2);
  unsigned short* kbuf   = (unsigned short*)carve((size_t)NN * DD * 2);
  unsigned short* vbuf   = (unsigned short*)carve((size_t)NN * DD * 2);
  unsigned short* abuf   = (unsigned short*)carve((size_t)NN * DD * 2);
  unsigned short* upbuf  = (unsigned short*)carve((size_t)NN * FF * 2);

  dim3 blk(256);
  // weight conversion/transpose
  for (int l = 0; l < LL; ++l) {
    dim3 gDD(DD / 32, DD / 32);
    transpose_f32_to_bf16<<<gDD, blk, 0, stream>>>(Wq + (size_t)l * DD * DD, WqT + (size_t)l * DD * DD, DD, DD);
    transpose_f32_to_bf16<<<gDD, blk, 0, stream>>>(Wk + (size_t)l * DD * DD, WkT + (size_t)l * DD * DD, DD, DD);
    transpose_f32_to_bf16<<<gDD, blk, 0, stream>>>(Wv + (size_t)l * DD * DD, WvT + (size_t)l * DD * DD, DD, DD);
    transpose_f32_to_bf16<<<gDD, blk, 0, stream>>>(Wo + (size_t)l * DD * DD, WoT + (size_t)l * DD * DD, DD, DD);
    transpose_f32_to_bf16<<<dim3(FF / 32, DD / 32), blk, 0, stream>>>(
        Wup + (size_t)l * DD * FF, WupT + (size_t)l * DD * FF, DD, FF);
    transpose_f32_to_bf16<<<dim3(DD / 32, FF / 32), blk, 0, stream>>>(
        Wdown + (size_t)l * DD * FF, WdownT + (size_t)l * DD * FF, FF, DD);
  }
  transpose_f32_to_bf16<<<dim3(VV / 32, DD / 32), blk, 0, stream>>>(Wlm, WlmT, DD, VV);

  // embeddings
  embed_kernel<<<NN, blk, 0, stream>>>(ids, tok, pos, xbuf);

  for (int l = 0; l < LL; ++l) {
    layernorm_bf16<<<NN, blk, 0, stream>>>(xbuf, ln1_s + (size_t)l * DD, ln1_b + (size_t)l * DD, hbuf);
    dim3 gQ(DD / 128, NN / 128);
    gemm_bf16<1><<<gQ, blk, 0, stream>>>(hbuf, WqT + (size_t)l * DD * DD, bq + (size_t)l * DD,
                                         nullptr, qbuf, NN, DD, DD);
    gemm_bf16<1><<<gQ, blk, 0, stream>>>(hbuf, WkT + (size_t)l * DD * DD, bk + (size_t)l * DD,
                                         nullptr, kbuf, NN, DD, DD);
    gemm_bf16<1><<<gQ, blk, 0, stream>>>(hbuf, WvT + (size_t)l * DD * DD, bv + (size_t)l * DD,
                                         nullptr, vbuf, NN, DD, DD);
    attention_kernel<<<dim3(HH, NN / BSB), blk, 0, stream>>>(qbuf, kbuf, vbuf, abuf);
    gemm_bf16<3><<<gQ, blk, 0, stream>>>(abuf, WoT + (size_t)l * DD * DD, bo + (size_t)l * DD,
                                         xbuf, nullptr, NN, DD, DD);
    layernorm_bf16<<<NN, blk, 0, stream>>>(xbuf, ln2_s + (size_t)l * DD, ln2_b + (size_t)l * DD, hbuf);
    gemm_bf16<2><<<dim3(FF / 128, NN / 128), blk, 0, stream>>>(
        hbuf, WupT + (size_t)l * DD * FF, bup + (size_t)l * FF, nullptr, upbuf, NN, FF, DD);
    gemm_bf16<3><<<dim3(DD / 128, NN / 128), blk, 0, stream>>>(
        upbuf, WdownT + (size_t)l * DD * FF, bdown + (size_t)l * DD, xbuf, nullptr, NN, DD, FF);
  }

  layernorm_bf16<<<NN, blk, 0, stream>>>(xbuf, fin_s, fin_b, hbuf);
  gemm_bf16<0><<<dim3(VV / 128, NN / 128), blk, 0, stream>>>(
      hbuf, WlmT, blm, (float*)d_out, nullptr, NN, VV, DD);
}
